// SwinMAE_76312978915976
// MI455X (gfx1250) — matrix operand order repaired and checked by reading, not compiled, not run on hardware
//
#include <hip/hip_runtime.h>
#include <math.h>

// ---------------------------------------------------------------------------
// CDNA5 (gfx1250) SwinMAE-3D forward. wave32, WMMA bf16 16x16x32, f32 accum.
// GEMM: 64x64 tile, 2 WMMA/wave (shared A frag), register-pipelined loads.
// ---------------------------------------------------------------------------

typedef __attribute__((ext_vector_type(16))) __bf16 v16bf;
typedef __attribute__((ext_vector_type(8)))  __bf16 v8bf;
typedef __attribute__((ext_vector_type(8)))  float  v8f;

#define GDIM 56
#define EDIM 96
#define LTOK (GDIM*GDIM*GDIM)            /* 175616 */
#define DMASK 14
#define NWIN_MASK (DMASK*DMASK*DMASK)    /* 2744 */
#define NKEEP 686
#define IMGD 112

// ---------------- wave helpers (wave32) ----------------
__device__ inline float wred_sum(float v){
#pragma unroll
  for (int m = 16; m >= 1; m >>= 1) v += __shfl_xor(v, m, 32);
  return v;
}
__device__ inline float wred_max(float v){
#pragma unroll
  for (int m = 16; m >= 1; m >>= 1) v = fmaxf(v, __shfl_xor(v, m, 32));
  return v;
}
__device__ inline v16bf cat8(v8bf lo, v8bf hi){
  v16bf r;
#pragma unroll
  for (int i = 0; i < 8; ++i){ r[i] = lo[i]; r[i+8] = hi[i]; }
  return r;
}
__device__ inline v8bf cvt8(float4 a, float4 b, float s){
  v8bf r;
  r[0]=(__bf16)(a.x*s); r[1]=(__bf16)(a.y*s); r[2]=(__bf16)(a.z*s); r[3]=(__bf16)(a.w*s);
  r[4]=(__bf16)(b.x*s); r[5]=(__bf16)(b.y*s); r[6]=(__bf16)(b.z*s); r[7]=(__bf16)(b.w*s);
  return r;
}
__device__ inline v8bf zero8(){
  v8bf r;
#pragma unroll
  for (int i = 0; i < 8; ++i) r[i] = (__bf16)0.f;
  return r;
}
__device__ inline v16bf cvt16(float4 a, float4 b, float4 c, float4 d, float s){
  return cat8(cvt8(a, b, s), cvt8(c, d, s));
}
__device__ inline v16bf zero16(){
  return cat8(zero8(), zero8());
}
__device__ inline v8f wmma_bf16(v16bf a, v16bf b, v8f c){
  return __builtin_amdgcn_wmma_f32_16x16x32_bf16(false, a, false, b, (short)0, c, false, false);
}

// ---------------------------------------------------------------------------
// Generic GEMM:  Y[M,N] = act( X[M,K] * W[N,K]^T + bias ) (+resid)
// 256 thr = 8 waves in 4(M)x2(N); wave computes 16x32 via 2 WMMA (shared A).
// K is a multiple of 32 in every call. M,N guarded.
// Software pipeline: next K-step's global loads issued before this step's WMMAs.
// ---------------------------------------------------------------------------
#define BM 64
#define BN 64
#define BK 32
__global__ __launch_bounds__(256) void k_gemm(
    const float* __restrict__ X, const float* __restrict__ W,
    const float* __restrict__ bias, const float* __restrict__ resid,
    float* __restrict__ Y, int M, int N, int K, int fuse_gelu)
{
  __shared__ __bf16 As[BM][BK];
  __shared__ __bf16 Bs[BN][BK];
  const int tid  = threadIdx.x;
  const int lane = tid & 31, wave = tid >> 5;
  const int wm = wave >> 1, wn = wave & 1;
  const int bm0 = blockIdx.y * BM, bn0 = blockIdx.x * BN;

  // staging slots: one 8-float row segment per thread for each of A and B
  const int ar = tid >> 2;            // 0..63
  const int ac = (tid & 3) * 8;       // 0,8,16,24
  const int gA = bm0 + ar;            // A source row
  const int gB = bn0 + ar;            // B source row
  const float* pA = &X[(size_t)(gA < M ? gA : 0) * K + ac];
  const float* pB = &W[(size_t)(gB < N ? gB : 0) * K + ac];
  const bool vA = (gA < M), vB = (gB < N);

  v8f acc0 = {0.f,0.f,0.f,0.f,0.f,0.f,0.f,0.f};
  v8f acc1 = {0.f,0.f,0.f,0.f,0.f,0.f,0.f,0.f};

  // prologue loads (K >= 32 always)
  float4 a0, a1, b0, b1;
  if (vA) { const float4* p = (const float4*)pA; a0 = p[0]; a1 = p[1]; }
  else    { a0 = a1 = make_float4(0.f,0.f,0.f,0.f); }
  if (vB) { const float4* p = (const float4*)pB; b0 = p[0]; b1 = p[1]; }
  else    { b0 = b1 = make_float4(0.f,0.f,0.f,0.f); }

  for (int k0 = 0; k0 < K; k0 += BK) {
    *(v8bf*)&As[ar][ac] = cvt8(a0, a1, 1.f);
    *(v8bf*)&Bs[ar][ac] = cvt8(b0, b1, 1.f);
    __syncthreads();

    const int kn = k0 + BK;
    if (kn < K) {
      if (vA) { const float4* p = (const float4*)(pA + kn); a0 = p[0]; a1 = p[1]; }
      if (vB) { const float4* p = (const float4*)(pB + kn); b0 = p[0]; b1 = p[1]; }
      if (kn + BK < K) __builtin_prefetch(pA + kn + BK, 0, 1);
    }

    const int mrow = wm*16 + (lane & 15);
    const int kA   = (lane >> 4) * 16;
    v16bf a = cat8(*(const v8bf*)&As[mrow][kA], *(const v8bf*)&As[mrow][kA + 8]);
    const int nr0 = wn*32 + (lane & 15);
    const int kB  = (lane >> 4) * 16;
    v16bf bb0 = cat8(*(const v8bf*)&Bs[nr0][kB],      *(const v8bf*)&Bs[nr0][kB + 8]);
    v16bf bb1 = cat8(*(const v8bf*)&Bs[nr0 + 16][kB], *(const v8bf*)&Bs[nr0 + 16][kB + 8]);
    acc0 = wmma_bf16(a, bb0, acc0);
    acc1 = wmma_bf16(a, bb1, acc1);
    __syncthreads();
  }

  // epilogue: two 16-col tiles per wave
#pragma unroll
  for (int half = 0; half < 2; ++half) {
    const int n = bn0 + wn*32 + half*16 + (lane & 15);
    const v8f& acc = half ? acc1 : acc0;
    if (n < N) {
      float bv = bias ? bias[n] : 0.f;
#pragma unroll
      for (int j = 0; j < 8; ++j) {
        int m = bm0 + wm*16 + ((lane >> 4) * 8 + j);
        if (m < M) {
          float v = acc[j] + bv;
          if (fuse_gelu) v = 0.5f * v * (1.f + erff(v * 0.70710678118654752f));
          if (resid) v += resid[(size_t)m * N + n];
          Y[(size_t)m * N + n] = v;
        }
      }
    }
  }
}

// ---------------------------------------------------------------------------
// LayerNorm over last dim (one-pass sum/sum-of-squares stats), 4 rows per 128-thread block.
// ---------------------------------------------------------------------------
__global__ __launch_bounds__(128) void k_ln(
    const float* __restrict__ Xin, float* __restrict__ Y,
    const float* __restrict__ g, const float* __restrict__ b, int M, int C)
{
  const int lane = threadIdx.x & 31, wave = threadIdx.x >> 5;
  const int row = blockIdx.x * 4 + wave;
  if (row >= M) return;
  const float* xr = Xin + (size_t)row * C;
  float s = 0.f, s2 = 0.f;
  for (int c = lane; c < C; c += 32) { float v = xr[c]; s += v; s2 += v*v; }
  s = wred_sum(s); s2 = wred_sum(s2);
  const float mean = s / C;
  const float inv  = rsqrtf(s2 / C - mean*mean + 1e-5f);
  float* yr = Y + (size_t)row * C;
  for (int c = lane; c < C; c += 32)
    yr[c] = (xr[c] - mean) * inv * g[c] + b[c];
}

// ---------------------------------------------------------------------------
// Fused windowed attention, one block per (window, head). T=343 pad 352, hd=32.
// ---------------------------------------------------------------------------
__device__ inline int swin_reg(int x, int D){ return x < D-7 ? 0 : (x < D-3 ? 1 : 2); }

__global__ __launch_bounds__(256) void k_attn(
    const float* __restrict__ QKV, float* __restrict__ O,
    const float* __restrict__ rpb, int heads, int dim, int Dg, int shift)
{
  const int T_ = 343;
  const int wh = blockIdx.x;
  const int head = wh % heads, widx = wh / heads;
  const int n = Dg / 7;
  const int wd = widx / (n*n), whh = (widx / n) % n, wwc = widx % n;
  const int threeC = 3 * dim;
  const int lane = threadIdx.x & 31, wave = threadIdx.x >> 5;
  const size_t rowbase = (size_t)widx * T_;

  __shared__ __bf16 Vt[32][352];  // V transposed  (22.5KB)
  __shared__ float  Ss[16][352];  // scores        (22.5KB)
  __shared__ __bf16 Ps[16][352];  // probs bf16    (11.25KB)
  __shared__ float  Os[16][32];   // out tile      (2KB)

  // stage V (transposed): float4 global loads, b16 LDS scatters
  for (int i = threadIdx.x; i < 352*8; i += 256) {
    int t = i >> 3, d0 = (i & 7) * 4;
    float4 v;
    if (t < T_) v = *(const float4*)&QKV[(rowbase + t)*threeC + 2*dim + head*32 + d0];
    else        v = make_float4(0.f, 0.f, 0.f, 0.f);
    Vt[d0+0][t] = (__bf16)v.x;
    Vt[d0+1][t] = (__bf16)v.y;
    Vt[d0+2][t] = (__bf16)v.z;
    Vt[d0+3][t] = (__bf16)v.w;
  }
  __syncthreads();

  const float qscale = 0.17677669529663687f; // 1/sqrt(32)

  for (int qt = 0; qt < 22; ++qt) {
    // ---- per-wave Q fragment (reused for all K tiles) ----
    const int qrow = qt*16 + (lane & 15);
    const int kA = (lane >> 4) * 16;
    v16bf afrag;
    if (qrow < T_) {
      const float4* q0 = (const float4*)&QKV[(rowbase + qrow)*threeC + head*32 + kA];
      const float4* q1 = (const float4*)&QKV[(rowbase + qrow)*threeC + head*32 + kA + 8];
      afrag = cvt16(q0[0], q0[1], q1[0], q1[1], qscale);
    } else {
      afrag = zero16();
    }

    // ---- S = Q K^T tiles, waves split over 22 key tiles ----
    for (int kt = wave; kt < 22; kt += 8) {
      const int ktok = kt*16 + (lane & 15);
      const int kB = (lane >> 4) * 16;
      v16bf bfrag;
      if (ktok < T_) {
        const float4* kp = (const float4*)&QKV[(rowbase + ktok)*threeC + dim + head*32 + kB];
        bfrag = cvt16(kp[0], kp[1], kp[2], kp[3], 1.f);
      } else {
        bfrag = zero16();
      }
      v8f acc = {0.f,0.f,0.f,0.f,0.f,0.f,0.f,0.f};
      acc = wmma_bf16(afrag, bfrag, acc);

      const int ncol = lane & 15;
      const int kk = kt*16 + ncol;
#pragma unroll
      for (int j = 0; j < 8; ++j) {
        const int m = (lane >> 4) * 8 + j;
        const int q = qt*16 + m;
        float v;
        if (q < T_ && kk < T_) {
          v = acc[j];
          const int qa = q/49,  qb = (q/7)%7,  qc = q%7;
          const int ka = kk/49, kb = (kk/7)%7, kc = kk%7;
          const int ridx = ((qa-ka+6)*13 + (qb-kb+6))*13 + (qc-kc+6);
          v += rpb[(size_t)ridx * heads + head];
          if (shift) {
            int lq = (swin_reg(wd*7+qa,Dg)*3 + swin_reg(whh*7+qb,Dg))*3 + swin_reg(wwc*7+qc,Dg);
            int lk = (swin_reg(wd*7+ka,Dg)*3 + swin_reg(whh*7+kb,Dg))*3 + swin_reg(wwc*7+kc,Dg);
            if (lq != lk) v -= 100.f;
          }
        } else {
          v = -1e30f;
        }
        Ss[m][kt*16 + ncol] = v;
      }
    }
    __syncthreads();

    // ---- softmax: 2 rows per wave ----
    for (int m = wave*2; m < wave*2 + 2; ++m) {
      float mx = -1e30f;
      for (int c = lane; c < 352; c += 32) mx = fmaxf(mx, Ss[m][c]);
      mx = wred_max(mx);
      float sum = 0.f;
      for (int c = lane; c < 352; c += 32) { float e = __expf(Ss[m][c] - mx); Ss[m][c] = e; sum += e; }
      sum = wred_sum(sum);
      const float inv = 1.f / sum;
      for (int c = lane; c < 352; c += 32) Ps[m][c] = (__bf16)(Ss[m][c] * inv);
    }
    for (int i = threadIdx.x; i < 16*32; i += 256) ((float*)Os)[i] = 0.f;
    __syncthreads();

    // ---- O = P V : 2 n-tiles x 11 K-chunks over 8 waves, LDS atomics reduce
    {
      const int ntile = wave & 1;
      v8f oacc = {0.f,0.f,0.f,0.f,0.f,0.f,0.f,0.f};
      for (int kc = (wave >> 1); kc < 11; kc += 4) {
        const int m = lane & 15;
        const int kA2 = (lane >> 4) * 16;
        v16bf pa = cat8(*(const v8bf*)&Ps[m][kc*32 + kA2],
                        *(const v8bf*)&Ps[m][kc*32 + kA2 + 8]);
        const int col = ntile*16 + (lane & 15);
        const int kB2 = kc*32 + (lane >> 4) * 16;
        v16bf vb = cat8(*(const v8bf*)&Vt[col][kB2],
                        *(const v8bf*)&Vt[col][kB2 + 8]);
        oacc = wmma_bf16(pa, vb, oacc);
      }
      const int col = ntile*16 + (lane & 15);
#pragma unroll
      for (int j = 0; j < 8; ++j)
        atomicAdd(&Os[(lane >> 4) * 8 + j][col], oacc[j]);
    }
    __syncthreads();

    for (int i = threadIdx.x; i < 16*32; i += 256) {
      int m = i >> 5, d = i & 31;
      int q = qt*16 + m;
      if (q < T_) O[(rowbase + q)*dim + head*32 + d] = Os[m][d];
    }
    __syncthreads();
  }
}

// ---------------------------------------------------------------------------
// Window partition with cyclic shift (roll by -shift), and inverse (+residual)
// ---------------------------------------------------------------------------
__global__ __launch_bounds__(256) void k_part(
    const float* __restrict__ H, float* __restrict__ Pb, int Dg, int C, int shift)
{
  size_t i = (size_t)blockIdx.x * 256 + threadIdx.x;
  size_t tot = (size_t)Dg*Dg*Dg*C;
  if (i >= tot) return;
  int ch = (int)(i % C);
  size_t t = i / C;
  int tw = (int)(t % 343); int widx = (int)(t / 343);
  int n = Dg / 7;
  int a = tw/49, b = (tw/7)%7, c = tw%7;
  int wd = widx/(n*n), wh = (widx/n)%n, ww = widx%n;
  int g0 = (wd*7 + a + shift) % Dg;
  int g1 = (wh*7 + b + shift) % Dg;
  int g2 = (ww*7 + c + shift) % Dg;
  Pb[i] = H[(((size_t)g0*Dg + g1)*Dg + g2)*C + ch];
}

__global__ __launch_bounds__(256) void k_unpart_add(
    const float* __restrict__ Pb, float* __restrict__ X, int Dg, int C, int shift)
{
  size_t i = (size_t)blockIdx.x * 256 + threadIdx.x;
  size_t tot = (size_t)Dg*Dg*Dg*C;
  if (i >= tot) return;
  int ch = (int)(i % C);
  size_t t = i / C;
  int tw = (int)(t % 343); int widx = (int)(t / 343);
  int n = Dg / 7;
  int a = tw/49, b = (tw/7)%7, c = tw%7;
  int wd = widx/(n*n), wh = (widx/n)%n, ww = widx%n;
  int g0 = (wd*7 + a + shift) % Dg;
  int g1 = (wh*7 + b + shift) % Dg;
  int g2 = (ww*7 + c + shift) % Dg;
  X[(((size_t)g0*Dg + g1)*Dg + g2)*C + ch] += Pb[i];
}

// ---------------------------------------------------------------------------
// Patch merging concat (2x2x2 -> 8C), and patch-expand scatter
// ---------------------------------------------------------------------------
__global__ __launch_bounds__(256) void k_merge_cat(
    const float* __restrict__ X, float* __restrict__ Y, int Dg, int C)
{
  int Dh = Dg >> 1;
  size_t tot = (size_t)Dh*Dh*Dh * 8 * C;
  size_t i = (size_t)blockIdx.x * 256 + threadIdx.x;
  if (i >= tot) return;
  int c8 = (int)(i % (8*C));
  size_t r = i / (8*C);
  int q = c8 % C, p = c8 / C;
  int pi = (p >> 2) & 1, pj = (p >> 1) & 1, pk = p & 1;
  int w = (int)(r % Dh); int h = (int)((r / Dh) % Dh); int d = (int)(r / ((size_t)Dh*Dh));
  size_t src = (((size_t)(2*d+pi)*Dg + (2*h+pj))*Dg + (2*w+pk));
  Y[i] = X[src * C + q];
}

__global__ __launch_bounds__(256) void k_expand_scatter(
    const float* __restrict__ Hd, float* __restrict__ Y, int Dg, int Ch)
{
  int D2 = Dg * 2;
  size_t tot = (size_t)D2*D2*D2 * Ch;
  size_t i = (size_t)blockIdx.x * 256 + threadIdx.x;
  if (i >= tot) return;
  int ch = (int)(i % Ch);
  size_t r = i / Ch;
  int ow = (int)(r % D2); int oh = (int)((r / D2) % D2); int od = (int)(r / ((size_t)D2*D2));
  int d = od >> 1, a = od & 1;
  int h = oh >> 1, b = oh & 1;
  int w = ow >> 1, c = ow & 1;
  size_t src = ((size_t)d*Dg + h)*Dg + w;
  int col = ((a*2 + b)*2 + c) * Ch + ch;
  Y[i] = Hd[src * (8*(size_t)Ch) + col];
}

// ---------------------------------------------------------------------------
// Patch embedding (2^3 conv == tiny GEMM), masking, prediction head, loss
// ---------------------------------------------------------------------------
__global__ __launch_bounds__(96) void k_patch_embed(
    const float* __restrict__ x, const float* __restrict__ pw,
    const float* __restrict__ pb, float* __restrict__ out)
{
  __shared__ float patch[8];
  int t = blockIdx.x;
  int d = t/(GDIM*GDIM), h = (t/GDIM)%GDIM, w = t%GDIM;
  if (threadIdx.x < 8) {
    int p = (threadIdx.x >> 2) & 1, q = (threadIdx.x >> 1) & 1, r = threadIdx.x & 1;
    patch[threadIdx.x] =
      x[((size_t)(2*d+p)*IMGD + (2*h+q))*IMGD + (2*w+r)];
  }
  __syncthreads();
  int e = threadIdx.x;
  float s = pb[e];
#pragma unroll
  for (int p = 0; p < 8; ++p) s += patch[p] * pw[e*8 + p];
  out[(size_t)t*EDIM + e] = s;
}

__global__ __launch_bounds__(256) void k_rank(const float* __restrict__ noise,
                                              int* __restrict__ keep)
{
  int i = blockIdx.x * 256 + threadIdx.x;
  if (i >= NWIN_MASK) return;
  float ni = noise[i];
  int rank = 0;
  for (int j = 0; j < NWIN_MASK; ++j) {
    float nj = noise[j];
    rank += (nj < ni) || (nj == ni && j < i);
  }
  keep[i] = (rank < NKEEP) ? 1 : 0;
}

__global__ __launch_bounds__(256) void k_mask_apply(
    const float* __restrict__ Hin, const int* __restrict__ keep,
    const float* __restrict__ mtok, float* __restrict__ Xout,
    float* __restrict__ maskOut)
{
  size_t i = (size_t)blockIdx.x * 256 + threadIdx.x;
  if (i >= (size_t)LTOK * EDIM) return;
  int ch = (int)(i % EDIM);
  size_t t = i / EDIM;
  int a = (int)(t/(GDIM*GDIM)), b = (int)((t/GDIM)%GDIM), c = (int)(t%GDIM);
  int widx = ((a>>2)*DMASK + (b>>2))*DMASK + (c>>2);
  int kp = keep[widx];
  Xout[i] = kp ? Hin[i] : mtok[ch];
  if (ch == 0) maskOut[t] = kp ? 0.f : 1.f;
}

__global__ __launch_bounds__(256) void k_pred(
    const float* __restrict__ Xd, const float* __restrict__ pw,
    const float* __restrict__ pb, float* __restrict__ pred)
{
  __shared__ float xs[32][EDIM];
  int t0 = blockIdx.x * 32;
  for (int i = threadIdx.x; i < 32*EDIM; i += 256) {
    int r = i / EDIM, c = i % EDIM;
    int t = t0 + r;
    xs[r][c] = (t < LTOK) ? Xd[(size_t)t*EDIM + c] : 0.f;
  }
  __syncthreads();
  for (int i = threadIdx.x; i < 32*8; i += 256) {
    int r = i >> 3, j = i & 7;
    int t = t0 + r;
    if (t < LTOK) {
      float s = pb[j];
      for (int c = 0; c < EDIM; ++c) s += xs[r][c] * pw[j*EDIM + c];
      pred[(size_t)t*8 + j] = s;
    }
  }
}

__global__ __launch_bounds__(256) void k_loss(
    const float* __restrict__ x, const float* __restrict__ pred,
    const float* __restrict__ mask, float* __restrict__ sums)
{
  int t = blockIdx.x * 256 + threadIdx.x;
  float num = 0.f, den = 0.f;
  if (t < LTOK) {
    int d = t/(GDIM*GDIM), h = (t/GDIM)%GDIM, w = t%GDIM;
    float acc = 0.f;
#pragma unroll
    for (int j = 0; j < 8; ++j) {
      int p = (j >> 2) & 1, q = (j >> 1) & 1, r = j & 1;
      float tgt = x[((size_t)(2*d+p)*IMGD + (2*h+q))*IMGD + (2*w+r)];
      float df = pred[(size_t)t*8 + j] - tgt;
      acc += df * df;
    }
    float mk = mask[t];
    num = acc * 0.125f * mk;
    den = mk;
  }
  num = wred_sum(num);
  den = wred_sum(den);
  if ((threadIdx.x & 31) == 0) {
    atomicAdd(&sums[0], num);
    atomicAdd(&sums[1], den);
  }
}

__global__ void k_zero(float* p, int n){
  int i = blockIdx.x * 256 + threadIdx.x;
  if (i < n) p[i] = 0.f;
}
__global__ void k_final(const float* __restrict__ sums, float* __restrict__ out0){
  out0[0] = sums[0] / sums[1];
}

// ---------------------------------------------------------------------------
// Host orchestration
// ---------------------------------------------------------------------------
struct BlkP { const float *fc1_b,*fc1_w,*fc2_b,*fc2_w,*ln1_b,*ln1_g,*ln2_b,*ln2_g,
                          *proj_b,*proj_w,*qkv_b,*qkv_w,*rpb; };
struct ExtP { const float *ln_b,*ln_g,*w; };

extern "C" void kernel_launch(void* const* d_in, const int* in_sizes, int n_in,
                              void* d_out, int out_size, void* d_ws, size_t ws_size,
                              hipStream_t stream) {
  (void)in_sizes; (void)n_in; (void)out_size; (void)ws_size;
  const float* x_in  = (const float*)d_in[0];
  const float* noise = (const float*)d_in[1];

  // ---- walk params in JAX pytree order (dict keys alphabetical) ----
  int cur = 2;
  auto nxt = [&]() { return (const float*)d_in[cur++]; };
  auto readBlk = [&](BlkP& b) {
    b.fc1_b=nxt(); b.fc1_w=nxt(); b.fc2_b=nxt(); b.fc2_w=nxt();
    b.ln1_b=nxt(); b.ln1_g=nxt(); b.ln2_b=nxt(); b.ln2_g=nxt();
    b.proj_b=nxt(); b.proj_w=nxt(); b.qkv_b=nxt(); b.qkv_w=nxt(); b.rpb=nxt();
  };
  auto readExt = [&](ExtP& e) { e.ln_b=nxt(); e.ln_g=nxt(); e.w=nxt(); };

  static const int ENC_DEPTH[4] = {2,2,6,2};
  static const int DEC_DEPTH[3] = {6,2,2};
  static const int ENC_HEADS[4] = {3,6,12,24};
  static const int DEC_HEADS[3] = {12,6,3};

  BlkP dec_blk[3][6]; ExtP dec_exp[2];
  BlkP enc_blk[4][6]; ExtP enc_mrg[3];
  ExtP fexp;

  for (int s = 0; s < 3; ++s) {
    for (int j = 0; j < DEC_DEPTH[s]; ++j) readBlk(dec_blk[s][j]);
    if (s < 2) readExt(dec_exp[s]);
  }
  for (int s = 0; s < 4; ++s) {
    for (int j = 0; j < ENC_DEPTH[s]; ++j) readBlk(enc_blk[s][j]);
    if (s < 3) readExt(enc_mrg[s]);
  }
  readExt(fexp);
  const float* mask_token = nxt();
  const float* nu_b = nxt(); const float* nu_g = nxt();
  const float* pe_b = nxt(); const float* pe_ln_b = nxt(); const float* pe_ln_g = nxt();
  const float* pe_w = nxt();
  const float* pred_b = nxt(); const float* pred_w = nxt();

  // ---- workspace layout (floats) ----
  const size_t SZT = (size_t)LTOK * EDIM;           // 16.86M
  float* WS  = (float*)d_ws;
  float* Xb  = WS;                                  // current features
  float* Hb  = WS + SZT;                            // LN out / attn out
  float* Pb  = WS + 2*SZT;                          // partitioned / proj out
  float* QKV = WS + 3*SZT;                          // up to L*288
  float* HID = QKV + (size_t)LTOK*288;              // up to L*384
  float* SUM = HID + (size_t)LTOK*384;
  int*   KEEP= (int*)(SUM + 8);

  float* out_loss = (float*)d_out;
  float* out_pred = out_loss + 1;
  float* out_mask = out_pred + (size_t)LTOK*8;

  auto gemm = [&](const float* X, const float* W, const float* b,
                  const float* res, float* Y, int M, int N, int K, int gelu) {
    dim3 g((N + BN - 1)/BN, (M + BM - 1)/BM);
    k_gemm<<<g, 256, 0, stream>>>(X, W, b, res, Y, M, N, K, gelu);
  };
  auto ew = [&](size_t tot){ return dim3((unsigned)((tot + 255)/256)); };

  auto run_block = [&](int D, int C, int heads, const BlkP& b, int shift) {
    int M = D*D*D;
    size_t MC = (size_t)M*C;
    k_ln<<<dim3((M+3)/4), 128, 0, stream>>>(Xb, Hb, b.ln1_g, b.ln1_b, M, C);
    k_part<<<ew(MC), 256, 0, stream>>>(Hb, Pb, D, C, shift);
    gemm(Pb, b.qkv_w, b.qkv_b, nullptr, QKV, M, 3*C, C, 0);
    int n = D/7; int nW = n*n*n;
    k_attn<<<dim3(nW*heads), 256, 0, stream>>>(QKV, Hb, b.rpb, heads, C, D, shift);
    gemm(Hb, b.proj_w, b.proj_b, nullptr, Pb, M, C, C, 0);
    k_unpart_add<<<ew(MC), 256, 0, stream>>>(Pb, Xb, D, C, shift);
    k_ln<<<dim3((M+3)/4), 128, 0, stream>>>(Xb, Hb, b.ln2_g, b.ln2_b, M, C);
    gemm(Hb, b.fc1_w, b.fc1_b, nullptr, HID, M, 4*C, C, 1 /*gelu*/);
    gemm(HID, b.fc2_w, b.fc2_b, Xb, Xb, M, C, 4*C, 0);
  };

  // ---- patch embed + LN + masking ----
  k_patch_embed<<<dim3(LTOK), 96, 0, stream>>>(x_in, pe_w, pe_b, Pb);
  k_ln<<<dim3((LTOK+3)/4), 128, 0, stream>>>(Pb, Hb, pe_ln_g, pe_ln_b, LTOK, EDIM);
  k_rank<<<dim3((NWIN_MASK+255)/256), 256, 0, stream>>>(noise, KEEP);
  k_mask_apply<<<ew(SZT), 256, 0, stream>>>(Hb, KEEP, mask_token, Xb, out_mask);

  // ---- encoder ----
  int D = GDIM, C = EDIM;
  for (int s = 0; s < 4; ++s) {
    for (int j = 0; j < ENC_DEPTH[s]; ++j)
      run_block(D, C, ENC_HEADS[s], enc_blk[s][j], (D > 7 && (j & 1)) ? 3 : 0);
    if (s < 3) {
      int Dh = D/2, Mh = Dh*Dh*Dh;
      k_merge_cat<<<ew((size_t)Mh*8*C), 256, 0, stream>>>(Xb, HID, D, C);
      k_ln<<<dim3((Mh+3)/4), 128, 0, stream>>>(HID, HID, enc_mrg[s].ln_g, enc_mrg[s].ln_b, Mh, 8*C);
      gemm(HID, enc_mrg[s].w, nullptr, nullptr, Xb, Mh, 2*C, 8*C, 0);
      D = Dh; C *= 2;
    }
  }

  // ---- first expand: D=7,C=768 -> D=14,C=384 ----
  {
    int M = D*D*D;
    gemm(Xb, fexp.w, nullptr, nullptr, HID, M, 4*C, C, 0);
    int Ch = C/2, D2 = 2*D, M2 = D2*D2*D2;
    k_expand_scatter<<<ew((size_t)M2*Ch), 256, 0, stream>>>(HID, Pb, D, Ch);
    k_ln<<<dim3((M2+3)/4), 128, 0, stream>>>(Pb, Xb, fexp.ln_g, fexp.ln_b, M2, Ch);
    D = D2; C = Ch;
  }

  // ---- decoder ----
  for (int s = 0; s < 3; ++s) {
    for (int j = 0; j < DEC_DEPTH[s]; ++j)
      run_block(D, C, DEC_HEADS[s], dec_blk[s][j], (D > 7 && (j & 1)) ? 3 : 0);
    if (s < 2) {
      int M = D*D*D;
      gemm(Xb, dec_exp[s].w, nullptr, nullptr, HID, M, 4*C, C, 0);
      int Ch = C/2, D2 = 2*D, M2 = D2*D2*D2;
      k_expand_scatter<<<ew((size_t)M2*Ch), 256, 0, stream>>>(HID, Pb, D, Ch);
      k_ln<<<dim3((M2+3)/4), 128, 0, stream>>>(Pb, Xb, dec_exp[s].ln_g, dec_exp[s].ln_b, M2, Ch);
      D = D2; C = Ch;
    }
  }

  // ---- head: LN, pred, loss ----
  k_ln<<<dim3((LTOK+3)/4), 128, 0, stream>>>(Xb, Hb, nu_g, nu_b, LTOK, EDIM);
  k_pred<<<dim3((LTOK+31)/32), 256, 0, stream>>>(Hb, pred_w, pred_b, out_pred);
  k_zero<<<dim3(1), 256, 0, stream>>>(SUM, 8);
  k_loss<<<dim3((LTOK+255)/256), 256, 0, stream>>>(x_in, out_pred, out_mask, SUM);
  k_final<<<dim3(1), 1, 0, stream>>>(SUM, out_loss);
}
